// MegNetLayer_81844896792587
// MI455X (gfx1250) — matrix operand order repaired and checked by reading, not compiled, not run on hardware
//
#include <hip/hip_runtime.h>
#include <hip/hip_bf16.h>

// ---------------------------------------------------------------------------
// MegNet layer on gfx1250 (wave32, WMMA f32_16x16x32_f16).
// All GEMMs run on the matrix pipe; activations staged through LDS in the
// exact CDNA5 fragment layouts (ISA 7.12.2). Softplus uses the hardware
// transcendentals (v_exp_f32 / v_log_f32) branch-free so the hot loop stays
// WMMA/memory bound instead of VALU bound.
// ---------------------------------------------------------------------------

typedef __attribute__((ext_vector_type(16))) _Float16 v16h;
typedef __attribute__((ext_vector_type(8)))  float    v8f;

#define TB   128   // threads per block (4 waves)
#define ROWS 64    // 16 rows per wave * 4 waves

__device__ __forceinline__ float softplus_f(float x) {
    // softplus(x) = max(x,0) + log1p(exp(-|x|)); branch-free, HW transcendentals.
    // exp(-|x|) underflows to 0 for large |x| -> log(1) = 0, both tails exact.
    float e = __expf(-fabsf(x));          // v_exp_f32
    return fmaxf(x, 0.0f) + __logf(1.0f + e);  // v_log_f32
}

// A fragment, 16-bit 16x32 (MxK): lane l -> row M=l&15, g=l>>4.
// VGPR v<4 holds K = 8g+2v, 8g+2v+1 ; v>=4 holds K = 16+8g+2(v-4), +1.
__device__ __forceinline__ v16h load_a_frag(const _Float16* __restrict__ row,
                                            int k0, int g) {
    v16h a;
#pragma unroll
    for (int v = 0; v < 8; ++v) {
        int kl = (v < 4) ? (8 * g + 2 * v) : (16 + 8 * g + 2 * (v - 4));
        a[2 * v]     = row[k0 + kl];
        a[2 * v + 1] = row[k0 + kl + 1];
    }
    return a;
}

// B fragment, 16-bit 32x16 (KxN): lane -> column N=l&15, g=l>>4; element i<8 holds K=8g+i, i>=8 holds K=16+8g+(i-8).
__device__ __forceinline__ v16h load_b_frag(const _Float16* __restrict__ W,
                                            int ldw, int k0, int n0, int lane) {
    const int gb = lane >> 4;
    const _Float16* r = W + (size_t)k0 * ldw + n0 + (lane & 15);
    v16h b;
#pragma unroll
    for (int i = 0; i < 16; ++i) {
        int kl = (i < 8) ? (8 * gb + i) : (16 + 8 * gb + (i - 8));
        b[i] = r[(size_t)kl * ldw];
    }
    return b;
}

// One wave computes the full 3-layer MLP for a 16-row tile held in LDS.
//   X:  [16][Kin] f16 (destroyed: reused as layer-2 output scratch)
//   H:  [16][64]  f16 per-wave scratch
//   out[nt][j] = softplus(layer3) for n = nt*16 + (lane&15), m = j + 8*(lane>>4)
__device__ void mlp_16rows(_Float16* __restrict__ X, int Kin,
                           const _Float16* __restrict__ W1, const float* __restrict__ B1,
                           const _Float16* __restrict__ W2, const float* __restrict__ B2,
                           const _Float16* __restrict__ W3, const float* __restrict__ B3,
                           _Float16* __restrict__ H, int lane, float out[2][8]) {
    const int g = lane >> 4;
    const int n = lane & 15;
    const _Float16* arow = X + (size_t)(lane & 15) * Kin;

    // ---- layer 1: [16,Kin] x [Kin,64] ----
    for (int nt = 0; nt < 4; ++nt) {
        v8f acc = {};
        for (int k0 = 0; k0 < Kin; k0 += 32) {
            v16h a = load_a_frag(arow, k0, g);
            v16h b = load_b_frag(W1, 64, k0, nt * 16, lane);
            acc = __builtin_amdgcn_wmma_f32_16x16x32_f16(false, a, false, b,
                                                         (short)0, acc, false, false);
        }
        float bias = B1[nt * 16 + n];
#pragma unroll
        for (int j = 0; j < 8; ++j) {
            float y = softplus_f(acc[j] + bias);
            H[(size_t)(j + 8 * g) * 64 + nt * 16 + n] = (_Float16)y;
        }
    }
    asm volatile("s_wait_dscnt 0" ::: "memory");  // in-wave LDS RAW ordering

    // ---- layer 2: [16,64] x [64,64], output into X region ----
    _Float16* H2 = X;  // layer-1 inputs fully consumed; 16*Kin >= 16*64
    const _Float16* hrow = H + (size_t)(lane & 15) * 64;
    for (int nt = 0; nt < 4; ++nt) {
        v8f acc = {};
#pragma unroll
        for (int kt = 0; kt < 2; ++kt) {
            v16h a = load_a_frag(hrow, kt * 32, g);
            v16h b = load_b_frag(W2, 64, kt * 32, nt * 16, lane);
            acc = __builtin_amdgcn_wmma_f32_16x16x32_f16(false, a, false, b,
                                                         (short)0, acc, false, false);
        }
        float bias = B2[nt * 16 + n];
#pragma unroll
        for (int j = 0; j < 8; ++j) {
            float y = softplus_f(acc[j] + bias);
            H2[(size_t)(j + 8 * g) * 64 + nt * 16 + n] = (_Float16)y;
        }
    }
    asm volatile("s_wait_dscnt 0" ::: "memory");

    // ---- layer 3: [16,64] x [64,32] ----
    const _Float16* h2row = H2 + (size_t)(lane & 15) * 64;
    for (int nt = 0; nt < 2; ++nt) {
        v8f acc = {};
#pragma unroll
        for (int kt = 0; kt < 2; ++kt) {
            v16h a = load_a_frag(h2row, kt * 32, g);
            v16h b = load_b_frag(W3, 32, kt * 32, nt * 16, lane);
            acc = __builtin_amdgcn_wmma_f32_16x16x32_f16(false, a, false, b,
                                                         (short)0, acc, false, false);
        }
        float bias = B3[nt * 16 + n];
#pragma unroll
        for (int j = 0; j < 8; ++j) out[nt][j] = softplus_f(acc[j] + bias);
    }
}

// ---------------------------------------------------------------------------
// Kernel 1: edge MLP + scatter-sum into atoms + bond-mean partials
// ---------------------------------------------------------------------------
__global__ __launch_bounds__(TB)
void edge_mlp_kernel(const float* __restrict__ bonds, const int* __restrict__ ba1,
                     const int* __restrict__ ba2, const float* __restrict__ atoms,
                     const float* __restrict__ state,
                     const float* __restrict__ w1, const float* __restrict__ b1,
                     const float* __restrict__ w2, const float* __restrict__ b2,
                     const float* __restrict__ w3, const float* __restrict__ b3,
                     float* __restrict__ bonds_out, float* __restrict__ summed,
                     float* __restrict__ cnt, float* __restrict__ bsum,
                     int nBonds, int nAtoms) {
    __shared__ _Float16 sW1[128 * 64];
    __shared__ _Float16 sW2[64 * 64];
    __shared__ _Float16 sW3[64 * 32];
    __shared__ _Float16 sX[ROWS * 128];
    __shared__ _Float16 sH[ROWS * 64];
    __shared__ float    sBS[32];

    const int t = threadIdx.x;
    __builtin_prefetch(w1, 0, 3);
    __builtin_prefetch(w2, 0, 3);
    for (int i = t; i < 128 * 64; i += TB) sW1[i] = (_Float16)w1[i];
    for (int i = t; i < 64 * 64;  i += TB) sW2[i] = (_Float16)w2[i];
    for (int i = t; i < 64 * 32;  i += TB) sW3[i] = (_Float16)w3[i];
    if (t < 32) sBS[t] = 0.0f;

    const int bond0 = blockIdx.x * ROWS;
    // Gather/concat [atoms[a1] | atoms[a2] | bonds | state] -> f16 LDS tile
    for (int idx = t; idx < ROWS * 128; idx += TB) {
        int r = idx >> 7, c = idx & 127;
        int bond = bond0 + r;
        if (bond >= nBonds) bond = nBonds - 1;
        float v;
        if (c < 32)       v = atoms[(size_t)ba1[bond] * 32 + c];
        else if (c < 64)  v = atoms[(size_t)ba2[bond] * 32 + (c - 32)];
        else if (c < 96)  v = bonds[(size_t)bond * 32 + (c - 64)];
        else              v = state[c - 96];
        sX[idx] = (_Float16)v;
    }
    // scatter-mean denominator: one add per bond
    if (t < ROWS && bond0 + t < nBonds)
        unsafeAtomicAdd(&cnt[ba1[bond0 + t]], 1.0f);
    __syncthreads();

    const int wave = t >> 5, lane = t & 31;
    const int g = lane >> 4, n = lane & 15;
    _Float16* X = sX + (size_t)wave * 16 * 128;
    _Float16* H = sH + (size_t)wave * 16 * 64;
    float out[2][8];
    mlp_16rows(X, 128, sW1, b1, sW2, b2, sW3, b3, H, lane, out);

    for (int nt = 0; nt < 2; ++nt) {
        float partial = 0.0f;
#pragma unroll
        for (int j = 0; j < 8; ++j) {
            int m = j + 8 * g;
            int bond = bond0 + wave * 16 + m;
            if (bond < nBonds) {
                float val = out[nt][j];
                bonds_out[(size_t)bond * 32 + nt * 16 + n] = val;
                unsafeAtomicAdd(&summed[(size_t)ba1[bond] * 32 + nt * 16 + n], val);
                partial += val;
            }
        }
        atomicAdd(&sBS[nt * 16 + n], partial);  // ds_add_f32 block reduce
    }
    __syncthreads();
    if (t < 32) unsafeAtomicAdd(&bsum[t], sBS[t]);
}

// ---------------------------------------------------------------------------
// Kernel 2: atom MLP (inputs: scatter-mean | atoms | state) + atom-mean partials
// ---------------------------------------------------------------------------
__global__ __launch_bounds__(TB)
void atom_mlp_kernel(const float* __restrict__ summed, const float* __restrict__ cnt,
                     const float* __restrict__ atoms, const float* __restrict__ state,
                     const float* __restrict__ w1, const float* __restrict__ b1,
                     const float* __restrict__ w2, const float* __restrict__ b2,
                     const float* __restrict__ w3, const float* __restrict__ b3,
                     float* __restrict__ atoms_out, float* __restrict__ asum,
                     int nAtoms) {
    __shared__ _Float16 sW1[96 * 64];
    __shared__ _Float16 sW2[64 * 64];
    __shared__ _Float16 sW3[64 * 32];
    __shared__ _Float16 sX[ROWS * 96];
    __shared__ _Float16 sH[ROWS * 64];
    __shared__ float    sAS[32];

    const int t = threadIdx.x;
    for (int i = t; i < 96 * 64; i += TB) sW1[i] = (_Float16)w1[i];
    for (int i = t; i < 64 * 64; i += TB) sW2[i] = (_Float16)w2[i];
    for (int i = t; i < 64 * 32; i += TB) sW3[i] = (_Float16)w3[i];
    if (t < 32) sAS[t] = 0.0f;

    const int atom0 = blockIdx.x * ROWS;
    for (int idx = t; idx < ROWS * 96; idx += TB) {
        int r = idx / 96, c = idx - r * 96;
        int atom = atom0 + r;
        if (atom >= nAtoms) atom = nAtoms - 1;
        float v;
        if (c < 32)      v = summed[(size_t)atom * 32 + c] / cnt[atom];
        else if (c < 64) v = atoms[(size_t)atom * 32 + (c - 32)];
        else             v = state[c - 64];
        sX[(size_t)r * 96 + c] = (_Float16)v;
    }
    __syncthreads();

    const int wave = t >> 5, lane = t & 31;
    const int g = lane >> 4, n = lane & 15;
    _Float16* X = sX + (size_t)wave * 16 * 96;
    _Float16* H = sH + (size_t)wave * 16 * 64;
    float out[2][8];
    mlp_16rows(X, 96, sW1, b1, sW2, b2, sW3, b3, H, lane, out);

    for (int nt = 0; nt < 2; ++nt) {
        float partial = 0.0f;
#pragma unroll
        for (int j = 0; j < 8; ++j) {
            int m = j + 8 * g;
            int atom = atom0 + wave * 16 + m;
            if (atom < nAtoms) {
                float val = out[nt][j];
                atoms_out[(size_t)atom * 32 + nt * 16 + n] = val;
                partial += val;
            }
        }
        atomicAdd(&sAS[nt * 16 + n], partial);
    }
    __syncthreads();
    if (t < 32) unsafeAtomicAdd(&asum[t], sAS[t]);
}

// ---------------------------------------------------------------------------
// Kernel 3: global/state MLP, single block (negligible work)
// ---------------------------------------------------------------------------
__global__ __launch_bounds__(64)
void state_mlp_kernel(const float* __restrict__ bsum, const float* __restrict__ asum,
                      const float* __restrict__ state,
                      const float* __restrict__ w1, const float* __restrict__ b1,
                      const float* __restrict__ w2, const float* __restrict__ b2,
                      const float* __restrict__ w3, const float* __restrict__ b3,
                      float* __restrict__ state_out, int nBonds, int nAtoms) {
    __shared__ float u[96];
    __shared__ float h1[64];
    __shared__ float h2[64];
    const int t = threadIdx.x;
    if (t < 32) {
        u[t]      = bsum[t] / (float)nBonds;
        u[32 + t] = asum[t] / (float)nAtoms;
        u[64 + t] = state[t];
    }
    __syncthreads();
    float acc = b1[t];
    for (int k = 0; k < 96; ++k) acc += u[k] * w1[(size_t)k * 64 + t];
    h1[t] = softplus_f(acc);
    __syncthreads();
    acc = b2[t];
    for (int k = 0; k < 64; ++k) acc += h1[k] * w2[(size_t)k * 64 + t];
    h2[t] = softplus_f(acc);
    __syncthreads();
    if (t < 32) {
        float a3 = b3[t];
        for (int k = 0; k < 64; ++k) a3 += h2[k] * w3[(size_t)k * 32 + t];
        state_out[t] = softplus_f(a3);
    }
}

// ---------------------------------------------------------------------------
extern "C" void kernel_launch(void* const* d_in, const int* in_sizes, int n_in,
                              void* d_out, int out_size, void* d_ws, size_t ws_size,
                              hipStream_t stream) {
    const float* bonds = (const float*)d_in[0];
    const int*   ba1   = (const int*)d_in[1];
    const int*   ba2   = (const int*)d_in[2];
    const float* atoms = (const float*)d_in[3];
    const float* state = (const float*)d_in[4];
    const float* e_w1 = (const float*)d_in[5];  const float* e_b1 = (const float*)d_in[6];
    const float* e_w2 = (const float*)d_in[7];  const float* e_b2 = (const float*)d_in[8];
    const float* e_w3 = (const float*)d_in[9];  const float* e_b3 = (const float*)d_in[10];
    const float* v_w1 = (const float*)d_in[11]; const float* v_b1 = (const float*)d_in[12];
    const float* v_w2 = (const float*)d_in[13]; const float* v_b2 = (const float*)d_in[14];
    const float* v_w3 = (const float*)d_in[15]; const float* v_b3 = (const float*)d_in[16];
    const float* u_w1 = (const float*)d_in[17]; const float* u_b1 = (const float*)d_in[18];
    const float* u_w2 = (const float*)d_in[19]; const float* u_b2 = (const float*)d_in[20];
    const float* u_w3 = (const float*)d_in[21]; const float* u_b3 = (const float*)d_in[22];

    const int nBonds = in_sizes[0] / 32;
    const int nAtoms = in_sizes[3] / 32;

    float* bonds_out = (float*)d_out;
    float* atoms_out = bonds_out + (size_t)nBonds * 32;
    float* state_out = atoms_out + (size_t)nAtoms * 32;

    float* summed = (float*)d_ws;                       // [nAtoms,32]
    float* cnt    = summed + (size_t)nAtoms * 32;       // [nAtoms]
    float* bsum   = cnt + nAtoms;                       // [32]
    float* asum   = bsum + 32;                          // [32]
    size_t zbytes = ((size_t)nAtoms * 32 + nAtoms + 64) * sizeof(float);
    hipMemsetAsync(d_ws, 0, zbytes, stream);

    dim3 blk(TB);
    edge_mlp_kernel<<<dim3((nBonds + ROWS - 1) / ROWS), blk, 0, stream>>>(
        bonds, ba1, ba2, atoms, state,
        e_w1, e_b1, e_w2, e_b2, e_w3, e_b3,
        bonds_out, summed, cnt, bsum, nBonds, nAtoms);

    atom_mlp_kernel<<<dim3((nAtoms + ROWS - 1) / ROWS), blk, 0, stream>>>(
        summed, cnt, atoms, state,
        v_w1, v_b1, v_w2, v_b2, v_w3, v_b3,
        atoms_out, asum, nAtoms);

    state_mlp_kernel<<<dim3(1), dim3(64), 0, stream>>>(
        bsum, asum, state,
        u_w1, u_b1, u_w2, u_b2, u_w3, u_b3,
        state_out, nBonds, nAtoms);
}
